// VelocityCrossAttention_37383395344857
// MI455X (gfx1250) — matrix operand order repaired and checked by reading, not compiled, not run on hardware
//
#include <hip/hip_runtime.h>
#include <math.h>

// Problem constants (match the reference)
#define BB   4
#define TT   512
#define JJ   17
#define CC   256
#define HH   8
#define DD   32
#define DOUT 256
#define MROWS (BB * TT * JJ)   // 34816 token rows; divisible by 64

typedef __bf16 v16bf __attribute__((ext_vector_type(16)));
typedef float  v8f   __attribute__((ext_vector_type(8)));

struct __attribute__((aligned(16))) U4 { unsigned x, y, z, w; };
union Frag { U4 u[2]; v16bf v; };

__device__ __forceinline__ unsigned short f2b(float f) {
  union { float f; unsigned u; } c; c.f = f;
  unsigned r = c.u + 0x7fffu + ((c.u >> 16) & 1u);   // round-to-nearest-even
  return (unsigned short)(r >> 16);
}
__device__ __forceinline__ float b2f(unsigned short h) {
  union { unsigned u; float f; } c; c.u = ((unsigned)h) << 16; return c.f;
}

// ---------------------------------------------------------------------------
// Elementwise: f32 -> bf16 convert
// ---------------------------------------------------------------------------
__global__ void cvt_f32_bf16(const float* __restrict__ in,
                             unsigned short* __restrict__ out, int n) {
  int i = blockIdx.x * blockDim.x + threadIdx.x;
  if (i < n) out[i] = f2b(in[i]);
}

// ---------------------------------------------------------------------------
// vel = gelu_exact(y[t] - y[t-1]) with y[-1] := y[0]   (layout b,t,j,2C)
// ---------------------------------------------------------------------------
__global__ void vel_gelu(const unsigned short* __restrict__ y,
                         unsigned short* __restrict__ vel, int n) {
  int i = blockIdx.x * blockDim.x + threadIdx.x;
  if (i >= n) return;
  const int strideT = JJ * 2 * CC;       // 8704
  int t = (i / strideT) % TT;
  float cur  = b2f(y[i]);
  float prev = (t == 0) ? cur : b2f(y[i - strideT]);
  float v = cur - prev;
  float g = 0.5f * v * (1.0f + erff(v * 0.70710678118654752f));
  vel[i] = f2b(g);
}

// ---------------------------------------------------------------------------
// vT[b][h][j][d][t] = kv[b][t][j][256 + h*32 + d]  (bf16 gather-transpose)
// ---------------------------------------------------------------------------
__global__ void transpose_v(const unsigned short* __restrict__ kv,
                            unsigned short* __restrict__ vT, int n) {
  int i = blockIdx.x * blockDim.x + threadIdx.x;
  if (i >= n) return;
  int t  = i & (TT - 1);
  int d  = (i >> 9) & (DD - 1);
  int r  = i >> 14;                      // (b*8+h)*17 + j
  int j  = r % JJ;
  int bh = r / JJ;
  int h  = bh & (HH - 1);
  int b  = bh >> 3;
  size_t src = ((size_t)(b * TT + t) * JJ + j) * (2 * CC) + CC + h * DD + d;
  vT[i] = kv[src];
}

// ---------------------------------------------------------------------------
// Generic WMMA bf16 GEMM:  C[M,N] = A[M,K] * W[N,K]^T (+ bias)
// A row-major bf16, W row-major bf16 (PyTorch Linear weight layout).
// 128 threads = 4 waves; block tile 64x64; K-step 32; one wave = 16x64 strip.
// Tiles are streamed global->LDS with GLOBAL_LOAD_ASYNC_TO_LDS_B128 into
// ping-pong buffers, synchronized with s_wait_asynccnt (each wave issues
// exactly 4 async-load instructions per tile; async loads complete in order,
// so asynccnt<=4 retires the current tile while the next stays in flight).
// ---------------------------------------------------------------------------
#define PITCH 40                 // LDS row pitch in bf16 elems (80 B)
#define TILE_ELEMS (64 * PITCH)  // one 64x32 tile (padded) in bf16 elems
#define BUF_ELEMS  (2 * TILE_ELEMS)

__global__ __launch_bounds__(128) void gemm_bf16(
    const unsigned short* __restrict__ A,
    const unsigned short* __restrict__ W,
    const float* __restrict__ bias,
    unsigned short* __restrict__ outB,   // optional bf16 output
    float* __restrict__ outF,            // optional f32 output
    int M, int N, int K) {
  __shared__ __align__(16) unsigned short lds[2 * BUF_ELEMS];  // 20.5 KB

  const int tid  = threadIdx.x;
  const int wave = tid >> 5;
  const int lane = tid & 31;
  const int lrow = lane & 15;
  const int hi   = lane >> 4;
  const int m0 = blockIdx.y * 64;
  const int n0 = blockIdx.x * 64;

  const unsigned ldsBase = (unsigned)(size_t)(&lds[0]);
  // per-thread staging coordinates: vectors 2*tid and 2*tid+1 of 256
  const int row0 = tid >> 1;                 // v = 2*tid    -> row, seg
  const int seg0 = (tid & 1) * 2 + 0;        //   (2 segs per thread, adjacent)
  // simpler: thread handles (row = tid>>1, segs {2*(tid&1), 2*(tid&1)+1})
  // i.e. two adjacent 16B vectors = 32 contiguous bytes per matrix

  auto stage = [&](int buf, int kk) {
    const unsigned base = ldsBase + (unsigned)(buf * BUF_ELEMS * 2);
    #pragma unroll
    for (int s = 0; s < 2; ++s) {
      int seg = seg0 + s;
      unsigned aOff = base + (unsigned)((row0 * PITCH + seg * 8) * 2);
      unsigned wOff = aOff + (unsigned)(TILE_ELEMS * 2);
      unsigned long long ga =
          (unsigned long long)(size_t)&A[(size_t)(m0 + row0) * K + kk + seg * 8];
      unsigned long long gw =
          (unsigned long long)(size_t)&W[(size_t)(n0 + row0) * K + kk + seg * 8];
      asm volatile("global_load_async_to_lds_b128 %0, %1, off"
                   :: "v"(aOff), "v"(ga) : "memory");
      asm volatile("global_load_async_to_lds_b128 %0, %1, off"
                   :: "v"(wOff), "v"(gw) : "memory");
    }
  };

  v8f acc0 = {0.f,0.f,0.f,0.f,0.f,0.f,0.f,0.f};
  v8f acc1 = acc0, acc2 = acc0, acc3 = acc0;

  const int NT = K >> 5;                     // number of 32-wide K tiles
  stage(0, 0);                               // prologue: tile 0 -> buffer 0

  for (int it = 0; it < NT; ++it) {
    const int cur = it & 1;
    if (it + 1 < NT) {
      stage(cur ^ 1, (it + 1) * 32);         // stream next tile, other buffer
      asm volatile("s_wait_asynccnt 0x4" ::: "memory");  // current tile done
    } else {
      asm volatile("s_wait_asynccnt 0x0" ::: "memory");
    }
    __syncthreads();

    const unsigned short* As = &lds[cur * BUF_ELEMS];
    const unsigned short* Ws = As + TILE_ELEMS;

    // A fragment: 16x32, documented interleaved K layout
    Frag af;
    {
      int r = (wave * 16 + lrow) * PITCH;
      af.u[0] = *(const U4*)&As[r + hi * 8];
      af.u[1] = *(const U4*)&As[r + 16 + hi * 8];
    }
    // B fragments: column n = lane%16, same interleaved K layout as A
    v8f* accs[4] = { &acc0, &acc1, &acc2, &acc3 };
    #pragma unroll
    for (int j = 0; j < 4; ++j) {
      Frag bf;
      int r = (j * 16 + lrow) * PITCH + hi * 8;
      bf.u[0] = *(const U4*)&Ws[r];
      bf.u[1] = *(const U4*)&Ws[r + 16];
      *accs[j] = __builtin_amdgcn_wmma_f32_16x16x32_bf16(
          false, af.v, false, bf.v, (short)0, *accs[j], false, false);
    }
    __syncthreads();   // all waves done reading `cur` before it is re-staged
  }

  // epilogue: lane holds rows {i, i+8}[hi], col = n0 + j*16 + lane%16
  v8f accv[4] = { acc0, acc1, acc2, acc3 };
  #pragma unroll
  for (int j = 0; j < 4; ++j) {
    int col = n0 + j * 16 + lrow;
    float bv = bias ? bias[col] : 0.f;
    #pragma unroll
    for (int i = 0; i < 8; ++i) {
      int row = m0 + wave * 16 + i + hi * 8;
      float val = accv[j][i] + bv;
      if (outF) outF[(size_t)row * N + col] = val;
      if (outB) outB[(size_t)row * N + col] = f2b(val);
    }
  }
}

// ---------------------------------------------------------------------------
// Attention over time per (b,h,j): one wave per 16-query tile.
// S = (Q K^T) * scale -> exact softmax -> O = P V.  d=32 == WMMA K exactly.
// ---------------------------------------------------------------------------
__global__ __launch_bounds__(32) void attn_kernel(
    const unsigned short* __restrict__ q,    // (b,t,j,C)      col = h*32+d
    const unsigned short* __restrict__ kv,   // (b,t,j,2C)     k: col h*32+d
    const unsigned short* __restrict__ vT,   // (b,h,j,d,t)
    unsigned short* __restrict__ out,        // (b,t,j,C)
    float scale) {
  __shared__ __align__(16) float S[16 * TT];            // 32 KB
  __shared__ __align__(16) unsigned short P[16 * TT];   // 16 KB
  __shared__ float red[32];

  const int bid  = blockIdx.x;
  const int qt   = bid & 31;
  const int rem  = bid >> 5;
  const int j    = rem % JJ;
  const int bh   = rem / JJ;
  const int h    = bh & (HH - 1);
  const int b    = bh >> 3;
  const int lane = threadIdx.x;
  const int lrow = lane & 15;
  const int hi   = lane >> 4;

  // Q fragment (A-matrix 16x32)
  Frag qf;
  {
    int tq = qt * 16 + lrow;
    const unsigned short* base =
        q + ((size_t)(b * TT + tq) * JJ + j) * CC + h * DD;
    qf.u[0] = *(const U4*)(base + hi * 8);
    qf.u[1] = *(const U4*)(base + 16 + hi * 8);
  }

  // S tiles: one WMMA per 16 keys (K-dim = d = 32)
  for (int kt = 0; kt < TT / 16; ++kt) {
    Frag kf;
    int tk = kt * 16 + lrow;
    const unsigned short* base =
        kv + ((size_t)(b * TT + tk) * JJ + j) * (2 * CC) + h * DD;
    kf.u[0] = *(const U4*)(base + hi * 8);
    kf.u[1] = *(const U4*)(base + 16 + hi * 8);
    v8f z = {0.f,0.f,0.f,0.f,0.f,0.f,0.f,0.f};
    v8f s = __builtin_amdgcn_wmma_f32_16x16x32_bf16(
        false, qf.v, false, kf.v, (short)0, z, false, false);
    #pragma unroll
    for (int i = 0; i < 8; ++i)
      S[(i + hi * 8) * TT + kt * 16 + lrow] = s[i] * scale;
  }
  __syncthreads();

  // exact softmax: 2 lanes per row, each owns half the 512 columns
  {
    const int c0 = hi * (TT / 2);
    float mx = -3.4e38f;
    for (int c = 0; c < TT / 2; ++c) mx = fmaxf(mx, S[lrow * TT + c0 + c]);
    red[lane] = mx;
    __syncthreads();
    float m = fmaxf(red[lrow], red[16 + lrow]);
    __syncthreads();
    float sum = 0.f;
    for (int c = 0; c < TT / 2; ++c) {
      float e = __expf(S[lrow * TT + c0 + c] - m);
      S[lrow * TT + c0 + c] = e;
      sum += e;
    }
    red[lane] = sum;
    __syncthreads();
    float inv = 1.0f / (red[lrow] + red[16 + lrow]);
    for (int c = 0; c < TT / 2; ++c)
      P[lrow * TT + c0 + c] = f2b(S[lrow * TT + c0 + c] * inv);
  }
  __syncthreads();

  // O = P (16xT) x V (Txd): accumulate over 16 chunks of 32 keys
  v8f o0 = {0.f,0.f,0.f,0.f,0.f,0.f,0.f,0.f};
  v8f o1 = o0;
  const unsigned short* vbase = vT + ((size_t)(b * HH + h) * JJ + j) * DD * TT;
  for (int c = 0; c < TT / 32; ++c) {
    Frag pf, v0, v1;
    pf.u[0] = *(const U4*)&P[lrow * TT + c * 32 + hi * 8];
    pf.u[1] = *(const U4*)&P[lrow * TT + c * 32 + 16 + hi * 8];
    const unsigned short* r0 = vbase + (size_t)lrow * TT + c * 32 + hi * 8;
    const unsigned short* r1 = vbase + (size_t)(16 + lrow) * TT + c * 32 + hi * 8;
    v0.u[0] = *(const U4*)(r0); v0.u[1] = *(const U4*)(r0 + 16);
    v1.u[0] = *(const U4*)(r1); v1.u[1] = *(const U4*)(r1 + 16);
    o0 = __builtin_amdgcn_wmma_f32_16x16x32_bf16(
        false, pf.v, false, v0.v, (short)0, o0, false, false);
    o1 = __builtin_amdgcn_wmma_f32_16x16x32_bf16(
        false, pf.v, false, v1.v, (short)0, o1, false, false);
  }

  #pragma unroll
  for (int i = 0; i < 8; ++i) {
    int t = qt * 16 + i + hi * 8;
    size_t basei = ((size_t)(b * TT + t) * JJ + j) * CC + h * DD;
    out[basei + lrow]      = f2b(o0[i]);
    out[basei + 16 + lrow] = f2b(o1[i]);
  }
}

// ---------------------------------------------------------------------------
// Launch
// ---------------------------------------------------------------------------
extern "C" void kernel_launch(void* const* d_in, const int* in_sizes, int n_in,
                              void* d_out, int out_size, void* d_ws, size_t ws_size,
                              hipStream_t stream) {
  const float* x   = (const float*)d_in[0];
  const float* Wq  = (const float*)d_in[1];
  const float* Wkv = (const float*)d_in[2];
  const float* Wc1 = (const float*)d_in[3];
  const float* bc1 = (const float*)d_in[4];
  const float* Wc2 = (const float*)d_in[5];
  const float* bc2 = (const float*)d_in[6];
  const float* Wp  = (const float*)d_in[7];
  const float* bp  = (const float*)d_in[8];
  float* out = (float*)d_out;

  char* ws = (char*)d_ws;
  size_t off = 0;
  auto alloc = [&](size_t elems) -> unsigned short* {
    unsigned short* p = (unsigned short*)(ws + off);
    off += ((elems * 2 + 255) / 256) * 256;
    return p;
  };
  const size_t MC  = (size_t)MROWS * CC;        // 8,912,896
  const size_t M2C = (size_t)MROWS * 2 * CC;    // 17,825,792
  unsigned short* xb    = alloc(MC);
  unsigned short* wc1b  = alloc(2 * CC * CC);
  unsigned short* wc2b  = alloc(CC * 2 * CC);
  unsigned short* wqb   = alloc(CC * CC);
  unsigned short* wkvb  = alloc(2 * CC * CC);
  unsigned short* wpb   = alloc(DOUT * CC);
  unsigned short* yb    = alloc(M2C);           // y; later reused as attn out
  unsigned short* velb  = alloc(M2C);
  unsigned short* velcb = alloc(MC);
  unsigned short* qb    = alloc(MC);
  unsigned short* kvb   = alloc(M2C);
  unsigned short* vTb   = alloc(MC);            // (b,h,j,d,t) = M*C elems

  auto cvt = [&](const float* src, unsigned short* dst, int n) {
    cvt_f32_bf16<<<(n + 255) / 256, 256, 0, stream>>>(src, dst, n);
  };
  cvt(x,   xb,   (int)MC);
  cvt(Wc1, wc1b, 2 * CC * CC);
  cvt(Wc2, wc2b, CC * 2 * CC);
  cvt(Wq,  wqb,  CC * CC);
  cvt(Wkv, wkvb, 2 * CC * CC);
  cvt(Wp,  wpb,  DOUT * CC);

  // 1) y = x @ Wc1^T + bc1            (M x 512, K=256)
  gemm_bf16<<<dim3(512 / 64, MROWS / 64), 128, 0, stream>>>(
      xb, wc1b, bc1, yb, nullptr, MROWS, 512, 256);
  // 2) vel = gelu(diff_t(y))
  vel_gelu<<<((int)M2C + 255) / 256, 256, 0, stream>>>(yb, velb, (int)M2C);
  // 3) velc = vel @ Wc2^T + bc2       (M x 256, K=512)
  gemm_bf16<<<dim3(256 / 64, MROWS / 64), 128, 0, stream>>>(
      velb, wc2b, bc2, velcb, nullptr, MROWS, 256, 512);
  // 4) q = velc @ Wq^T                (M x 256, K=256)
  gemm_bf16<<<dim3(256 / 64, MROWS / 64), 128, 0, stream>>>(
      velcb, wqb, nullptr, qb, nullptr, MROWS, 256, 256);
  // 5) kv = x @ Wkv^T                 (M x 512, K=256)
  gemm_bf16<<<dim3(512 / 64, MROWS / 64), 128, 0, stream>>>(
      xb, wkvb, nullptr, kvb, nullptr, MROWS, 512, 256);
  // 6) vT gather-transpose
  transpose_v<<<((int)MC + 255) / 256, 256, 0, stream>>>(kvb, vTb, (int)MC);
  // 7) attention (reuse yb as bf16 attn output, (b,t,j,C))
  const float scale = 0.17677669529663688f;     // 32^-0.5
  attn_kernel<<<BB * HH * JJ * (TT / 16), 32, 0, stream>>>(
      qb, kvb, vTb, yb, scale);
  // 8) final = attn_out @ Wp^T + bp -> f32 d_out
  gemm_bf16<<<dim3(DOUT / 64, MROWS / 64), 128, 0, stream>>>(
      yb, wpb, bp, nullptr, out, MROWS, DOUT, 256);
}
